// SimplifiedLlama4Attention_88459146428653
// MI455X (gfx1250) — compile-verified, matrix operand order checked by reading, not run on hardware
//
#include <hip/hip_runtime.h>
#include <hip/hip_bf16.h>
#include <math.h>

// ---- problem constants -----------------------------------------------------
#define HID 4096
#define NH  32
#define NKV 8
#define HD  128
#define SEQ 2048

typedef __attribute__((ext_vector_type(16))) __bf16 v16bf;
typedef __attribute__((ext_vector_type(8)))  float  v8f;

union Frag16 { unsigned u[8]; v16bf v; };   // build fragments via b32 loads

// round-to-nearest-even f32 -> bf16
__device__ __forceinline__ __bf16 to_bf16(float f) {
  unsigned u = __float_as_uint(f);
  unsigned r = u + 0x7FFFu + ((u >> 16) & 1u);
  unsigned short h = (unsigned short)(r >> 16);
  __bf16 b; __builtin_memcpy(&b, &h, 2); return b;
}
__device__ __forceinline__ __bf16 bits_to_bf16(unsigned short h) {
  __bf16 b; __builtin_memcpy(&b, &h, 2); return b;
}
// pack two f32 into one dword of two RNE bf16 (lo, hi)
__device__ __forceinline__ unsigned pack2_bf16(float x, float y) {
  unsigned ux = __float_as_uint(x);
  ux = (ux + 0x7FFFu + ((ux >> 16) & 1u)) >> 16;
  unsigned uy = __float_as_uint(y);
  uy = (uy + 0x7FFFu + ((uy >> 16) & 1u)) & 0xFFFF0000u;
  return ux | uy;
}

// k index pattern for 16-bit A/B fragments (ISA 05_wmma §7.12.2),
// element pair p (covers elements 2p, 2p+1), lane-half given separately.
__device__ __forceinline__ int frag_k(int p, int half) {
  return 2 * (p & 3) + 8 * half + 16 * (p >> 2);
}

#define WMMA_BF16(a, b, c) \
  __builtin_amdgcn_wmma_f32_16x16x32_bf16(false, (a), false, (b), (short)0, (c), false, false)

// ---- elementwise f32 -> bf16 ----------------------------------------------
__global__ void cvt_f32_bf16(const float* __restrict__ src,
                             __bf16* __restrict__ dst, int n) {
  int i = blockIdx.x * blockDim.x + threadIdx.x;
  if (i < n) dst[i] = to_bf16(src[i]);
}

// ---- V repack: f32 [k, c] -> bf16 pair-interleaved [(k/2), c, k&1] ---------
__global__ void pack_v_bf16(const float* __restrict__ src,
                            __bf16* __restrict__ dst, int n) {
  int i = blockIdx.x * blockDim.x + threadIdx.x;
  if (i < n) {
    const int k = i >> 10;            // key row   (ld = NKV*HD = 1024)
    const int c = i & 1023;           // column
    dst[(size_t)(k >> 1) * 2048 + c * 2 + (k & 1)] = to_bf16(src[i]);
  }
}

// ---- generic bf16 GEMM: C[M,N] = A[M,K] * B[K,N], f32 accumulate ----------
// 256 threads = 8 waves; block tile 64(M) x 128(N); double-buffered LDS,
// K steps of 32. Wave tile 32x32: 2 A-frags x 2 B-frags -> 4 wmma / step.
#define LDP 40   // LDS row pitch (bf16 elems): 80B rows, 16B-aligned, spreads banks
__global__ __launch_bounds__(256) void gemm_bf16_nn(
    const __bf16* __restrict__ A, const __bf16* __restrict__ B,
    float* __restrict__ C, int M, int N, int K, int lda, int ldb, int ldc)
{
  __shared__ __align__(16) __bf16 As[2][64  * LDP];   // [m][k]
  __shared__ __align__(16) __bf16 Bt[2][128 * LDP];   // [n][k] (transposed)

  const int tid  = threadIdx.x;
  const int lane = tid & 31;
  const int wid  = tid >> 5;
  const int wm   = wid & 1;             // wave M tile 0..1 (32 rows)
  const int wn   = wid >> 1;            // wave N tile 0..3 (32 cols)
  const int half = lane >> 4;
  const int ml   = lane & 15;
  const int bm   = blockIdx.y * 64;
  const int bn   = blockIdx.x * 128;

  // global->LDS coverage: A 64x32 (8/thread), B 32x128 (16/thread)
  const int ar = tid >> 2, ac = (tid & 3) * 8;
  const int kr = tid >> 3, kc = (tid & 7) * 16;

  const __bf16* gA = A + (size_t)(bm + ar) * lda + ac;
  const __bf16* gB = B + (size_t)kr * ldb + bn + kc;

  auto stage = [&](int buf, uint4 av, uint4 bv0, uint4 bv1) {
    *(uint4*)(&As[buf][ar * LDP + ac]) = av;
    unsigned w0[4] = {bv0.x, bv0.y, bv0.z, bv0.w};
    unsigned w1[4] = {bv1.x, bv1.y, bv1.z, bv1.w};
#pragma unroll
    for (int j = 0; j < 4; ++j) {
      Bt[buf][(kc + 2 * j)         * LDP + kr] = bits_to_bf16((unsigned short)(w0[j] & 0xFFFF));
      Bt[buf][(kc + 2 * j + 1)     * LDP + kr] = bits_to_bf16((unsigned short)(w0[j] >> 16));
      Bt[buf][(kc + 8 + 2 * j)     * LDP + kr] = bits_to_bf16((unsigned short)(w1[j] & 0xFFFF));
      Bt[buf][(kc + 8 + 2 * j + 1) * LDP + kr] = bits_to_bf16((unsigned short)(w1[j] >> 16));
    }
  };

  v8f c00 = {}, c01 = {}, c10 = {}, c11 = {};

  // prologue: stage K-step 0
  {
    uint4 av  = *(const uint4*)(gA);
    uint4 bv0 = *(const uint4*)(gB);
    uint4 bv1 = *(const uint4*)(gB + 8);
    stage(0, av, bv0, bv1);
  }
  __syncthreads();

  for (int k0 = 0; k0 < K; k0 += 32) {
    const int cur  = (k0 >> 5) & 1;
    const bool more = (k0 + 32 < K);
    uint4 nav, nbv0, nbv1;
    if (more) {   // issue next-step global loads before compute (latency hiding)
      nav  = *(const uint4*)(gA + k0 + 32);
      nbv0 = *(const uint4*)(gB + (size_t)(k0 + 32) * ldb);
      nbv1 = *(const uint4*)(gB + (size_t)(k0 + 32) * ldb + 8);
    }

    const __bf16* Asc = As[cur];
    const __bf16* Btc = Bt[cur];
    Frag16 a0, a1, b0, b1;
#pragma unroll
    for (int p = 0; p < 8; ++p) {
      const int kk = frag_k(p, half);
      a0.u[p] = *(const unsigned*)(Asc + (wm * 32 + ml)      * LDP + kk);
      a1.u[p] = *(const unsigned*)(Asc + (wm * 32 + 16 + ml) * LDP + kk);
      b0.u[p] = *(const unsigned*)(Btc + (wn * 32 + ml)      * LDP + kk);
      b1.u[p] = *(const unsigned*)(Btc + (wn * 32 + 16 + ml) * LDP + kk);
    }
    c00 = WMMA_BF16(a0.v, b0.v, c00);
    c01 = WMMA_BF16(a0.v, b1.v, c01);
    c10 = WMMA_BF16(a1.v, b0.v, c10);
    c11 = WMMA_BF16(a1.v, b1.v, c11);

    if (more) {
      __syncthreads();                 // readers of buf[cur^1] (2 steps ago) done
      stage(cur ^ 1, nav, nbv0, nbv1);
      __syncthreads();                 // staged data visible for next step
    }
  }

  const int m0 = bm + wm * 32 + 8 * half;
  float* c = C + (size_t)m0 * ldc + bn + wn * 32 + ml;
#pragma unroll
  for (int r = 0; r < 8; ++r) {
    c[(size_t)r * ldc]             = c00[r];
    c[(size_t)r * ldc + 16]        = c01[r];
    c[(size_t)(r + 16) * ldc]      = c10[r];
    c[(size_t)(r + 16) * ldc + 16] = c11[r];
  }
}

// ---- RoPE + L2 norm, one wave per (head, position) -------------------------
__global__ __launch_bounds__(32) void rope_l2norm(
    const float* __restrict__ X, __bf16* __restrict__ Y, int nheads)
{
  const int h    = blockIdx.x;
  const int m    = blockIdx.y;
  const int lane = threadIdx.x;
  const float* x = X + (size_t)m * nheads * HD + h * HD;
  __bf16*      y = Y + (size_t)m * nheads * HD + h * HD;

  const float kLn = 0.1025184670182755f;   // ln(500000)/128
  float o[4];
  float ss = 0.f;
#pragma unroll
  for (int i = 0; i < 2; ++i) {
    const int p = lane + i * 32;
    const int d = 2 * p;
    const float inv = __expf(-(float)d * kLn);
    const float fr  = (float)m * inv;
    const float c = __cosf(fr), s = __sinf(fr);
    const float xr = x[d], xi = x[d + 1];
    const float orr = xr * c - xi * s;
    const float oii = xr * s + xi * c;
    o[2 * i] = orr; o[2 * i + 1] = oii;
    ss += orr * orr + oii * oii;
  }
#pragma unroll
  for (int off = 16; off > 0; off >>= 1) ss += __shfl_xor(ss, off, 32);
  const float rms = rsqrtf(ss * (1.0f / 128.0f) + 1e-6f);
#pragma unroll
  for (int i = 0; i < 2; ++i) {
    const int d = 2 * (lane + i * 32);
    y[d]     = to_bf16(o[2 * i] * rms);
    y[d + 1] = to_bf16(o[2 * i + 1] * rms);
  }
}

// ---- scores = scale * Q Kᵀ : one wave -> 16(m) x 64(n); A reused x4 --------
__global__ __launch_bounds__(32) void attn_logits(
    const __bf16* __restrict__ Qbf, const __bf16* __restrict__ Kbf,
    float* __restrict__ P)
{
  const int bx = blockIdx.x;           // key group of 64
  const int bm = blockIdx.y;           // query tile of 16
  const int h  = blockIdx.z;
  if (bx * 64 > bm * 16 + 15) return;  // fully above diagonal (wave-uniform)
  const int lane = threadIdx.x & 31;
  const int half = lane >> 4, ml = lane & 15;
  const int kv = h >> 2;

  const __bf16* qrow  = Qbf + (size_t)(bm * 16 + ml) * (NH * HD) + h * HD;
  const __bf16* kbase = Kbf + (size_t)(bx * 64 + ml) * (NKV * HD) + kv * HD;

  v8f acc[4] = {{}, {}, {}, {}};
#pragma unroll
  for (int k0 = 0; k0 < HD; k0 += 32) {
    Frag16 a;
#pragma unroll
    for (int p = 0; p < 8; ++p)
      a.u[p] = *(const unsigned*)(qrow + k0 + frag_k(p, half));
#pragma unroll
    for (int s = 0; s < 4; ++s) {
      const __bf16* krow = kbase + (size_t)s * 16 * (NKV * HD);
      Frag16 b;
#pragma unroll
      for (int p = 0; p < 8; ++p)
        b.u[p] = *(const unsigned*)(krow + k0 + frag_k(p, half));
      acc[s] = WMMA_BF16(a.v, b.v, acc[s]);
    }
  }
  const float scale = 0.08838834764831845f;   // 1/sqrt(128)
  float* out = P + (size_t)h * SEQ * SEQ
                 + (size_t)(bm * 16 + 8 * half) * SEQ + bx * 64 + ml;
#pragma unroll
  for (int s = 0; s < 4; ++s)
#pragma unroll
    for (int r = 0; r < 8; ++r)
      out[(size_t)r * SEQ + s * 16] = acc[s][r] * scale;
}

// ---- causal softmax over one row, in place in d_out ------------------------
__global__ __launch_bounds__(256) void softmax_causal(float* __restrict__ P)
{
  const int m = blockIdx.x;
  const int h = blockIdx.y;
  float* row = P + ((size_t)h * SEQ + m) * SEQ;
  __shared__ float red[256];
  const int t = threadIdx.x;

  float vals[8];
  float mx = -3.0e38f;
#pragma unroll
  for (int i = 0; i < 8; ++i) {
    const int j = t + i * 256;
    vals[i] = row[j];
    if (j <= m) mx = fmaxf(mx, vals[i]);
  }
  red[t] = mx; __syncthreads();
  for (int s = 128; s > 0; s >>= 1) {
    if (t < s) red[t] = fmaxf(red[t], red[t + s]);
    __syncthreads();
  }
  mx = red[0]; __syncthreads();

  float sum = 0.f;
#pragma unroll
  for (int i = 0; i < 8; ++i) {
    const int j = t + i * 256;
    if (j <= m) sum += __expf(vals[i] - mx);
  }
  red[t] = sum; __syncthreads();
  for (int s = 128; s > 0; s >>= 1) {
    if (t < s) red[t] += red[t + s];
    __syncthreads();
  }
  const float inv = 1.0f / red[0];
#pragma unroll
  for (int i = 0; i < 8; ++i) {
    const int j = t + i * 256;
    row[j] = (j <= m) ? __expf(vals[i] - mx) * inv : 0.0f;
  }
}

// ---- O = P V : one wave -> 16(m) x 128(full HD); A reused x8 ---------------
// V is pair-interleaved: Vpk[(k/2)*2048 + c*2 + (k&1)]
__global__ __launch_bounds__(32) void attn_pv(
    const float* __restrict__ P, const __bf16* __restrict__ Vpk,
    __bf16* __restrict__ AO)
{
  const int bm = blockIdx.x;           // query tile of 16
  const int h  = blockIdx.y;
  const int lane = threadIdx.x & 31;
  const int half = lane >> 4, ml = lane & 15;
  const int kv = h >> 2;

  const float* prow = P + (size_t)h * SEQ * SEQ + (size_t)(bm * 16 + ml) * SEQ;

  v8f acc[8] = {{}, {}, {}, {}, {}, {}, {}, {}};
  const int kmax = bm * 16 + 16;       // probs beyond the row are exactly 0
  for (int k0 = 0; k0 < kmax; k0 += 32) {
    Frag16 a;
#pragma unroll
    for (int p = 0; p < 8; ++p) {
      const float2 pv = *(const float2*)(prow + k0 + frag_k(p, half));
      a.u[p] = pack2_bf16(pv.x, pv.y);
    }
#pragma unroll
    for (int s = 0; s < 8; ++s) {
      const int c = kv * HD + s * 16 + ml;     // V column for this sub-tile
      Frag16 b;
#pragma unroll
      for (int p = 0; p < 8; ++p) {
        const int kk = k0 + frag_k(p, half);   // even
        b.u[p] = *(const unsigned*)(Vpk + (size_t)(kk >> 1) * 2048 + c * 2);
      }
      acc[s] = WMMA_BF16(a.v, b.v, acc[s]);
    }
  }
  __bf16* o = AO + (size_t)(bm * 16 + 8 * half) * (NH * HD) + h * HD + ml;
#pragma unroll
  for (int s = 0; s < 8; ++s)
#pragma unroll
    for (int r = 0; r < 8; ++r)
      o[(size_t)r * (NH * HD) + s * 16] = to_bf16(acc[s][r]);
}

// ---- host driver -----------------------------------------------------------
extern "C" void kernel_launch(void* const* d_in, const int* in_sizes, int n_in,
                              void* d_out, int out_size, void* d_ws, size_t ws_size,
                              hipStream_t stream)
{
  const float* hs = (const float*)d_in[0];
  // d_in[1] attention_mask: unused (causal handled by index -> exact zeros)
  // d_in[2] position_ids:   unused (positions == row index for B=1)
  const float* Wq = (const float*)d_in[3];
  const float* Wk = (const float*)d_in[4];
  const float* Wv = (const float*)d_in[5];
  const float* Wo = (const float*)d_in[6];

  float* out   = (float*)d_out;                       // [2048, 4096] f32
  float* attnW = out + (size_t)SEQ * HID;             // [32, 2048, 2048] f32

  char* w = (char*)d_ws;
  auto alloc = [&](size_t bytes) -> char* {
    char* p = w; w += (bytes + 255) & ~(size_t)255; return p;
  };
  __bf16* Xbf = (__bf16*)alloc((size_t)SEQ * HID * 2);
  __bf16* Wqb = (__bf16*)alloc((size_t)HID * NH  * HD * 2);
  __bf16* Wkb = (__bf16*)alloc((size_t)HID * NKV * HD * 2);
  __bf16* Wvb = (__bf16*)alloc((size_t)HID * NKV * HD * 2);
  __bf16* Wob = (__bf16*)alloc((size_t)NH * HD * HID * 2);
  float*  Qf  = (float*) alloc((size_t)SEQ * NH  * HD * 4);
  float*  Kf  = (float*) alloc((size_t)SEQ * NKV * HD * 4);
  float*  Vf  = (float*) alloc((size_t)SEQ * NKV * HD * 4);
  __bf16* Qbf = (__bf16*)alloc((size_t)SEQ * NH  * HD * 2);
  __bf16* Kbf = (__bf16*)alloc((size_t)SEQ * NKV * HD * 2);
  __bf16* Vpk = (__bf16*)alloc((size_t)SEQ * NKV * HD * 2);
  __bf16* AOb = (__bf16*)alloc((size_t)SEQ * NH  * HD * 2);

  auto cvt = [&](const float* s, __bf16* d, int n) {
    cvt_f32_bf16<<<(n + 255) / 256, 256, 0, stream>>>(s, d, n);
  };
  cvt(hs, Xbf, SEQ * HID);
  cvt(Wq, Wqb, HID * NH * HD);
  cvt(Wk, Wkb, HID * NKV * HD);
  cvt(Wv, Wvb, HID * NKV * HD);
  cvt(Wo, Wob, NH * HD * HID);

  // QKV projections (block tile 64x128)
  gemm_bf16_nn<<<dim3((NH * HD) / 128, SEQ / 64), 256, 0, stream>>>(
      Xbf, Wqb, Qf, SEQ, NH * HD, HID, HID, NH * HD, NH * HD);
  gemm_bf16_nn<<<dim3((NKV * HD) / 128, SEQ / 64), 256, 0, stream>>>(
      Xbf, Wkb, Kf, SEQ, NKV * HD, HID, HID, NKV * HD, NKV * HD);
  gemm_bf16_nn<<<dim3((NKV * HD) / 128, SEQ / 64), 256, 0, stream>>>(
      Xbf, Wvb, Vf, SEQ, NKV * HD, HID, HID, NKV * HD, NKV * HD);

  // RoPE + L2 norm (fused); V pair-interleave repack
  rope_l2norm<<<dim3(NH,  SEQ), 32, 0, stream>>>(Qf, Qbf, NH);
  rope_l2norm<<<dim3(NKV, SEQ), 32, 0, stream>>>(Kf, Kbf, NKV);
  pack_v_bf16<<<(SEQ * NKV * HD + 255) / 256, 256, 0, stream>>>(
      Vf, Vpk, SEQ * NKV * HD);

  // attention: logits (lower-triangular tiles) -> softmax in place -> PV
  attn_logits<<<dim3(SEQ / 64, SEQ / 16, NH), 32, 0, stream>>>(Qbf, Kbf, attnW);
  softmax_causal<<<dim3(SEQ, NH), 256, 0, stream>>>(attnW);
  attn_pv<<<dim3(SEQ / 16, NH), 32, 0, stream>>>(attnW, Vpk, AOb);

  // output projection
  gemm_bf16_nn<<<dim3(HID / 128, SEQ / 64), 256, 0, stream>>>(
      AOb, Wob, out, SEQ, HID, NH * HD, NH * HD, HID, HID);
}
